// RMHA_20246475833971
// MI455X (gfx1250) — matrix operand order repaired and checked by reading, not compiled, not run on hardware
//
#include <hip/hip_runtime.h>
#include <cstdint>

#define QL 2048
#define ML 2048
#define KL 4096
#define BS 2
#define NH 16
#define DH 64
#define DM 1024

#define AS1 __attribute__((address_space(1)))
#define AS3 __attribute__((address_space(3)))

typedef __attribute__((ext_vector_type(16))) __bf16 v16bf;
typedef __attribute__((ext_vector_type(8)))  float  v8f;
typedef __attribute__((ext_vector_type(4)))  int    v4i;

static __device__ inline v8f wmma_bf16(v16bf a, v16bf b, v8f c) {
  return __builtin_amdgcn_wmma_f32_16x16x32_bf16(false, a, false, b, (short)0, c,
                                                 false, false);
}

// ---- async copy 16B global -> LDS (ASYNCcnt-tracked) ----------------------
static __device__ inline void async_cp16(const __bf16* gp, __bf16* lp) {
#if __has_builtin(__builtin_amdgcn_global_load_async_to_lds_b128)
  __builtin_amdgcn_global_load_async_to_lds_b128(
      (AS1 v4i*)(uintptr_t)gp, (AS3 v4i*)(uintptr_t)lp, 0, 0);
#else
  unsigned int loff = (unsigned int)(uintptr_t)lp;
  unsigned long long ga = (unsigned long long)(uintptr_t)gp;
  asm volatile("global_load_async_to_lds_b128 %0, %1, off"
               :: "v"(loff), "v"(ga) : "memory");
#endif
}
static __device__ inline void wait_async1() {
#if __has_builtin(__builtin_amdgcn_s_wait_asynccnt)
  __builtin_amdgcn_s_wait_asynccnt(1);
#else
  asm volatile("s_wait_asynccnt 0x1" ::: "memory");
#endif
}
static __device__ inline void wait_async0() {
#if __has_builtin(__builtin_amdgcn_s_wait_asynccnt)
  __builtin_amdgcn_s_wait_asynccnt(0);
#else
  asm volatile("s_wait_asynccnt 0x0" ::: "memory");
#endif
}

// A-matrix fragment (16x32 bf16): lane = row; lane-group g holds K chunks
// {8g..8g+7} and {16+8g..23+8g}  (ISA 7.12.2, 16-bit A 16x32 layout)
static __device__ inline v16bf a_frag_f32(const float* rp, int g) {
  v16bf r;
#pragma unroll
  for (int t = 0; t < 8; ++t) {
    r[t]     = (__bf16)rp[8 * g + t];
    r[t + 8] = (__bf16)rp[16 + 8 * g + t];
  }
  return r;
}
static __device__ inline v16bf a_frag_bf16(const __bf16* rp, int g) {
  v16bf r;
#pragma unroll
  for (int t = 0; t < 8; ++t) {
    r[t]     = rp[8 * g + t];
    r[t + 8] = rp[16 + 8 * g + t];
  }
  return r;
}
// B-matrix fragment (32x16 bf16): lane = column (rp = source row of the
// column); lane-group g holds K chunks {8g..8g+7} and {16+8g..23+8g}
static __device__ inline v16bf b_frag_bf16(const __bf16* rp, int g) {
  v16bf r;
#pragma unroll
  for (int t = 0; t < 8; ++t) {
    r[t]     = rp[8 * g + t];
    r[t + 8] = rp[16 + 8 * g + t];
  }
  return r;
}

// ---------------------------------------------------------------- transpose
__global__ __launch_bounds__(256) void k_transpose_bf16(const float* __restrict__ src,
                                                        __bf16* __restrict__ dst,
                                                        int R, int C) {
  __shared__ float tile[32][33];
  int c0 = blockIdx.x * 32, r0 = blockIdx.y * 32;
  int tx = threadIdx.x & 31, ty = threadIdx.x >> 5;
#pragma unroll
  for (int i = ty; i < 32; i += 8) tile[i][tx] = src[(size_t)(r0 + i) * C + c0 + tx];
  __syncthreads();
#pragma unroll
  for (int i = ty; i < 32; i += 8)
    dst[(size_t)(c0 + i) * R + r0 + tx] = (__bf16)tile[tx][i];
}

// ---------------------------------------------------------------- qkv GEMM
// C[8192 x 3072] = cat[8192 x 1024] @ Wqkv; B tile staged to LDS with async
// copies (double buffered); epilogue scatters into Qu/Qv/Kh/Vt.
__global__ __launch_bounds__(256) void k_qkv(const float* __restrict__ x,
                                             const float* __restrict__ mem,
                                             const __bf16* __restrict__ Wt,
                                             const float* __restrict__ ub,
                                             const float* __restrict__ vb,
                                             __bf16* __restrict__ Qu,
                                             __bf16* __restrict__ Qv,
                                             __bf16* __restrict__ Kh,
                                             __bf16* __restrict__ Vt) {
  __shared__ __bf16 bbuf[2][64 * 32];
  const int lane = threadIdx.x & 31, g = lane >> 4, ln = lane & 15;
  const int w = threadIdx.x >> 5;
  const int mw = blockIdx.x * 128 + (w & 3) * 32;
  const int nwl = (w >> 2) * 32;
  const int nw = blockIdx.y * 64 + nwl;

  const float* arow[2];
#pragma unroll
  for (int ms = 0; ms < 2; ++ms) {
    int r = mw + 16 * ms + ln;
    arow[ms] = (r < ML * BS) ? (mem + (size_t)r * DM) : (x + (size_t)(r - ML * BS) * DM);
  }
  // async B staging: thread copies 16B: col = tid>>2, chunk = tid&3
  const int ccol = threadIdx.x >> 2, cchk = threadIdx.x & 3;
  const __bf16* bsrc = Wt + (size_t)(blockIdx.y * 64 + ccol) * DM + cchk * 8;
  __bf16* bdst0 = &bbuf[0][ccol * 32 + cchk * 8];
  __bf16* bdst1 = &bbuf[1][ccol * 32 + cchk * 8];

  async_cp16(bsrc, bdst0);
  v8f acc[2][2] = {};
  for (int kt = 0; kt < 31; ++kt) {
    const int k0 = kt * 32;
    async_cp16(bsrc + k0 + 32, (kt & 1) ? bdst0 : bdst1);
    wait_async1();
    __syncthreads();
    const __bf16* bb = bbuf[kt & 1];
    v16bf a0 = a_frag_f32(arow[0] + k0, g);
    v16bf a1 = a_frag_f32(arow[1] + k0, g);
    v16bf b0 = b_frag_bf16(bb + (nwl + ln) * 32, g);
    v16bf b1 = b_frag_bf16(bb + (nwl + 16 + ln) * 32, g);
    acc[0][0] = wmma_bf16(a0, b0, acc[0][0]);
    acc[0][1] = wmma_bf16(a0, b1, acc[0][1]);
    acc[1][0] = wmma_bf16(a1, b0, acc[1][0]);
    acc[1][1] = wmma_bf16(a1, b1, acc[1][1]);
    __syncthreads();
  }
  {
    const int k0 = 31 * 32;
    wait_async0();
    __syncthreads();
    const __bf16* bb = bbuf[1];
    v16bf a0 = a_frag_f32(arow[0] + k0, g);
    v16bf a1 = a_frag_f32(arow[1] + k0, g);
    v16bf b0 = b_frag_bf16(bb + (nwl + ln) * 32, g);
    v16bf b1 = b_frag_bf16(bb + (nwl + 16 + ln) * 32, g);
    acc[0][0] = wmma_bf16(a0, b0, acc[0][0]);
    acc[0][1] = wmma_bf16(a0, b1, acc[0][1]);
    acc[1][0] = wmma_bf16(a1, b0, acc[1][0]);
    acc[1][1] = wmma_bf16(a1, b1, acc[1][1]);
  }
#pragma unroll
  for (int ms = 0; ms < 2; ++ms)
#pragma unroll
    for (int ns = 0; ns < 2; ++ns)
#pragma unroll
      for (int e = 0; e < 8; ++e) {
        float val = acc[ms][ns][e];
        int r = mw + 16 * ms + e + 8 * g;
        int c = nw + 16 * ns + ln;
        int t = r >> 1, bb2 = r & 1;
        if (c < DM) {                       // q: current segment only
          if (t >= ML) {
            int nh = c >> 6, dd = c & 63;
            size_t o = ((size_t)(bb2 * NH + nh) * QL + (t - ML)) * DH + dd;
            Qu[o] = (__bf16)(val + ub[c]);
            Qv[o] = (__bf16)(val + vb[c]);
          }
        } else if (c < 2 * DM) {            // k: row major [b][h][KL][DH]
          int cc = c - DM, nh = cc >> 6, dd = cc & 63;
          Kh[((size_t)(bb2 * NH + nh) * KL + t) * DH + dd] = (__bf16)val;
        } else {                            // v: transposed [b][h][DH][KL]
          int cc = c - 2 * DM, nh = cc >> 6, dd = cc & 63;
          Vt[((size_t)(bb2 * NH + nh) * DH + dd) * KL + t] = (__bf16)val;
        }
      }
}

// ---------------------------------------------------------------- R GEMM
static __device__ inline v16bf a_sincos(int m, int k0, int g) {
  v16bf r;
  float p = (float)(KL - 1 - m);
#pragma unroll
  for (int t = 0; t < 16; ++t) {
    int kk = (t < 8) ? (8 * g + t) : (16 + 8 * g + (t - 8));
    int k = k0 + kk;
    int j = k & 511;
    float f = __expf((float)j * -0.017988946f);   // -ln(10000)/512
    float ang = p * f;
    r[t] = (__bf16)((k < 512) ? __sinf(ang) : __cosf(ang));
  }
  return r;
}

__global__ __launch_bounds__(256) void k_rproj(const __bf16* __restrict__ Wrt,
                                               __bf16* __restrict__ Rh) {
  __shared__ __bf16 bbuf[2][64 * 32];
  const int lane = threadIdx.x & 31, g = lane >> 4, ln = lane & 15;
  const int w = threadIdx.x >> 5;
  const int mw = blockIdx.x * 128 + (w & 3) * 32;
  const int nwl = (w >> 2) * 32;
  const int nw = blockIdx.y * 64 + nwl;

  const int ccol = threadIdx.x >> 2, cchk = threadIdx.x & 3;
  const __bf16* bsrc = Wrt + (size_t)(blockIdx.y * 64 + ccol) * DM + cchk * 8;
  __bf16* bdst0 = &bbuf[0][ccol * 32 + cchk * 8];
  __bf16* bdst1 = &bbuf[1][ccol * 32 + cchk * 8];

  async_cp16(bsrc, bdst0);
  v8f acc[2][2] = {};
  for (int kt = 0; kt < 31; ++kt) {
    const int k0 = kt * 32;
    async_cp16(bsrc + k0 + 32, (kt & 1) ? bdst0 : bdst1);
    wait_async1();
    __syncthreads();
    const __bf16* bb = bbuf[kt & 1];
    v16bf a0 = a_sincos(mw + ln, k0, g);
    v16bf a1 = a_sincos(mw + 16 + ln, k0, g);
    v16bf b0 = b_frag_bf16(bb + (nwl + ln) * 32, g);
    v16bf b1 = b_frag_bf16(bb + (nwl + 16 + ln) * 32, g);
    acc[0][0] = wmma_bf16(a0, b0, acc[0][0]);
    acc[0][1] = wmma_bf16(a0, b1, acc[0][1]);
    acc[1][0] = wmma_bf16(a1, b0, acc[1][0]);
    acc[1][1] = wmma_bf16(a1, b1, acc[1][1]);
    __syncthreads();
  }
  {
    const int k0 = 31 * 32;
    wait_async0();
    __syncthreads();
    const __bf16* bb = bbuf[1];
    v16bf a0 = a_sincos(mw + ln, k0, g);
    v16bf a1 = a_sincos(mw + 16 + ln, k0, g);
    v16bf b0 = b_frag_bf16(bb + (nwl + ln) * 32, g);
    v16bf b1 = b_frag_bf16(bb + (nwl + 16 + ln) * 32, g);
    acc[0][0] = wmma_bf16(a0, b0, acc[0][0]);
    acc[0][1] = wmma_bf16(a0, b1, acc[0][1]);
    acc[1][0] = wmma_bf16(a1, b0, acc[1][0]);
    acc[1][1] = wmma_bf16(a1, b1, acc[1][1]);
  }
#pragma unroll
  for (int ms = 0; ms < 2; ++ms)
#pragma unroll
    for (int ns = 0; ns < 2; ++ns)
#pragma unroll
      for (int e = 0; e < 8; ++e) {
        int m = mw + 16 * ms + e + 8 * g;
        int c = nw + 16 * ns + ln;
        int nh = c >> 6, dd = c & 63;
        Rh[((size_t)nh * KL + m) * DH + dd] = (__bf16)acc[ms][ns][e];
      }
}

// ---------------------------------------------------------------- flash attn
// One wave per (b, head, 16-query tile). Online softmax. BD via 3 sliding
// 16-wide Qv@R^T windows + diagonal extraction with ds_bpermute shuffles.
__global__ __launch_bounds__(256) void k_flash(const __bf16* __restrict__ Qu,
                                               const __bf16* __restrict__ Qv,
                                               const __bf16* __restrict__ Kh,
                                               const __bf16* __restrict__ Vt,
                                               const __bf16* __restrict__ Rh,
                                               __bf16* __restrict__ AO) {
  __shared__ __bf16 pbuf[8][16 * 32];
  const int lane = threadIdx.x & 31, g = lane >> 4, ln = lane & 15;
  const int w = threadIdx.x >> 5;
  const int task = blockIdx.x * 8 + w;
  const int it = task & 127;
  const int n = (task >> 7) & 15;
  const int b = task >> 11;
  const int i0 = it * 16;

  const __bf16* qu = Qu + (size_t)(b * NH + n) * QL * DH;
  const __bf16* qv = Qv + (size_t)(b * NH + n) * QL * DH;
  const __bf16* kh = Kh + (size_t)(b * NH + n) * KL * DH;
  const __bf16* vt = Vt + (size_t)(b * NH + n) * DH * KL;
  const __bf16* rh = Rh + (size_t)n * KL * DH;

  const __bf16* qurow = qu + (size_t)(i0 + ln) * DH;
  const __bf16* qvrow = qv + (size_t)(i0 + ln) * DH;
  v16bf aqu0 = a_frag_bf16(qurow, g), aqu1 = a_frag_bf16(qurow + 32, g);
  v16bf aqv0 = a_frag_bf16(qvrow, g), aqv1 = a_frag_bf16(qvrow + 32, g);

  v8f acc[4] = {};
  float mrow[8], lrow[8];
#pragma unroll
  for (int e = 0; e < 8; ++e) { mrow[e] = -1e30f; lrow[e] = 0.f; }

  const float scale = 0.125f;               // 1/sqrt(64)
  const int kend = (i0 + ML + 16 + 31) & ~31;
  __bf16* pb = pbuf[w];

  for (int j0 = 0; j0 < kend; j0 += 32) {
    v8f zf = {};
    // prefetch next iteration's K/V/R tiles into caches
    if (j0 + 32 < kend) {
      const int jn = j0 + 32;
      __builtin_prefetch(kh + (size_t)(jn + lane) * DH, 0, 3);
      __builtin_prefetch(vt + (size_t)lane * KL + jn, 0, 3);
      __builtin_prefetch(vt + (size_t)(lane + 32) * KL + jn, 0, 3);
      int mn = jn - i0 + QL - 16 + lane;
      mn = mn > KL - 1 ? KL - 1 : mn;
      __builtin_prefetch(rh + (size_t)mn * DH, 0, 3);
    }
    // ---- AC = (q+u) @ K^T, two 16-col sub-tiles
    const __bf16* k0r = kh + (size_t)(j0 + ln) * DH;
    const __bf16* k1r = kh + (size_t)(j0 + 16 + ln) * DH;
    v8f ac0 = wmma_bf16(aqu1, b_frag_bf16(k0r + 32, g),
                        wmma_bf16(aqu0, b_frag_bf16(k0r, g), zf));
    v8f ac1 = wmma_bf16(aqu1, b_frag_bf16(k1r + 32, g),
                        wmma_bf16(aqu0, b_frag_bf16(k1r, g), zf));
    // ---- BD windows: m = j - i + QL - 1 band, 3 x 16 columns
    const int mstart = j0 - i0 + QL - 16;
    v8f bd[3];
#pragma unroll
    for (int wi = 0; wi < 3; ++wi) {
      int m = mstart + wi * 16 + ln;
      m = m < 0 ? 0 : (m > KL - 1 ? KL - 1 : m);
      const __bf16* rr = rh + (size_t)m * DH;
      bd[wi] = wmma_bf16(aqv1, b_frag_bf16(rr + 32, g),
                         wmma_bf16(aqv0, b_frag_bf16(rr, g), zf));
    }
    // ---- extract diagonal band, mask, scale
    float s0[8], s1[8];
#pragma unroll
    for (int e = 0; e < 8; ++e) {
      int il = e + 8 * g;
      int delta = ln - il;
      int idxA = (delta + 15) & 15;
      int idxB = (delta - 1) & 15;
      float b0v = __shfl(bd[0][e], g * 16 + idxA, 32);
      float b1a = __shfl(bd[1][e], g * 16 + idxB, 32);
      float b1b = __shfl(bd[1][e], g * 16 + idxA, 32);
      float b2v = __shfl(bd[2][e], g * 16 + idxB, 32);
      float bdv0 = (delta > 0) ? b1a : b0v;
      float bdv1 = (delta > 0) ? b2v : b1b;
      int i = i0 + il;
      s0[e] = (j0 + ln <= i + ML) ? (ac0[e] + bdv0) * scale : -3.0e38f;
      s1[e] = (j0 + 16 + ln <= i + ML) ? (ac1[e] + bdv1) * scale : -3.0e38f;
    }
    // ---- online softmax (16-lane butterfly per row)
    float alpha[8];
#pragma unroll
    for (int e = 0; e < 8; ++e) {
      float mx = fmaxf(s0[e], s1[e]);
      mx = fmaxf(mx, __shfl_xor(mx, 1, 32));
      mx = fmaxf(mx, __shfl_xor(mx, 2, 32));
      mx = fmaxf(mx, __shfl_xor(mx, 4, 32));
      mx = fmaxf(mx, __shfl_xor(mx, 8, 32));
      float mnew = fmaxf(mrow[e], mx);
      float p0 = __expf(s0[e] - mnew);
      float p1 = __expf(s1[e] - mnew);
      float a = __expf(mrow[e] - mnew);
      float rs = p0 + p1;
      rs += __shfl_xor(rs, 1, 32);
      rs += __shfl_xor(rs, 2, 32);
      rs += __shfl_xor(rs, 4, 32);
      rs += __shfl_xor(rs, 8, 32);
      lrow[e] = lrow[e] * a + rs;
      mrow[e] = mnew;
      alpha[e] = a;
      pb[(e + 8 * g) * 32 + ln] = (__bf16)p0;       // stage P tile (D-layout)
      pb[(e + 8 * g) * 32 + 16 + ln] = (__bf16)p1;
    }
#pragma unroll
    for (int nt = 0; nt < 4; ++nt)
#pragma unroll
      for (int e = 0; e < 8; ++e) acc[nt][e] *= alpha[e];
    asm volatile("s_wait_dscnt 0" ::: "memory");    // wave-local LDS RAW
    // ---- reload P in A-layout, accumulate P @ V
    v16bf ap;
#pragma unroll
    for (int t = 0; t < 8; ++t) {
      ap[t]     = pb[ln * 32 + 8 * g + t];
      ap[t + 8] = pb[ln * 32 + 16 + 8 * g + t];
    }
#pragma unroll
    for (int nt = 0; nt < 4; ++nt) {
      const __bf16* vrow = vt + (size_t)(nt * 16 + ln) * KL + j0;
      acc[nt] = wmma_bf16(ap, b_frag_bf16(vrow, g), acc[nt]);
    }
  }
  // ---- normalize + store attention output (bf16, GEMM-A layout)
#pragma unroll
  for (int e = 0; e < 8; ++e) lrow[e] = 1.0f / lrow[e];
#pragma unroll
  for (int nt = 0; nt < 4; ++nt)
#pragma unroll
    for (int e = 0; e < 8; ++e) {
      int i = i0 + e + 8 * g;
      int d = nt * 16 + ln;
      AO[((size_t)i * BS + b) * DM + n * DH + d] = (__bf16)(acc[nt][e] * lrow[e]);
    }
}

// ---------------------------------------------------------------- out GEMM
__global__ __launch_bounds__(256) void k_out(const __bf16* __restrict__ AO,
                                             const __bf16* __restrict__ Wot,
                                             float* __restrict__ out) {
  __shared__ __bf16 bbuf[2][64 * 32];
  const int lane = threadIdx.x & 31, g = lane >> 4, ln = lane & 15;
  const int w = threadIdx.x >> 5;
  const int mw = blockIdx.x * 128 + (w & 3) * 32;
  const int nwl = (w >> 2) * 32;
  const int nw = blockIdx.y * 64 + nwl;

  const int ccol = threadIdx.x >> 2, cchk = threadIdx.x & 3;
  const __bf16* bsrc = Wot + (size_t)(blockIdx.y * 64 + ccol) * DM + cchk * 8;
  __bf16* bdst0 = &bbuf[0][ccol * 32 + cchk * 8];
  __bf16* bdst1 = &bbuf[1][ccol * 32 + cchk * 8];

  async_cp16(bsrc, bdst0);
  v8f acc[2][2] = {};
  for (int kt = 0; kt < 31; ++kt) {
    const int k0 = kt * 32;
    async_cp16(bsrc + k0 + 32, (kt & 1) ? bdst0 : bdst1);
    wait_async1();
    __syncthreads();
    const __bf16* bb = bbuf[kt & 1];
    v16bf a0 = a_frag_bf16(AO + (size_t)(mw + ln) * DM + k0, g);
    v16bf a1 = a_frag_bf16(AO + (size_t)(mw + 16 + ln) * DM + k0, g);
    v16bf b0 = b_frag_bf16(bb + (nwl + ln) * 32, g);
    v16bf b1 = b_frag_bf16(bb + (nwl + 16 + ln) * 32, g);
    acc[0][0] = wmma_bf16(a0, b0, acc[0][0]);
    acc[0][1] = wmma_bf16(a0, b1, acc[0][1]);
    acc[1][0] = wmma_bf16(a1, b0, acc[1][0]);
    acc[1][1] = wmma_bf16(a1, b1, acc[1][1]);
    __syncthreads();
  }
  {
    const int k0 = 31 * 32;
    wait_async0();
    __syncthreads();
    const __bf16* bb = bbuf[1];
    v16bf a0 = a_frag_bf16(AO + (size_t)(mw + ln) * DM + k0, g);
    v16bf a1 = a_frag_bf16(AO + (size_t)(mw + 16 + ln) * DM + k0, g);
    v16bf b0 = b_frag_bf16(bb + (nwl + ln) * 32, g);
    v16bf b1 = b_frag_bf16(bb + (nwl + 16 + ln) * 32, g);
    acc[0][0] = wmma_bf16(a0, b0, acc[0][0]);
    acc[0][1] = wmma_bf16(a0, b1, acc[0][1]);
    acc[1][0] = wmma_bf16(a1, b0, acc[1][0]);
    acc[1][1] = wmma_bf16(a1, b1, acc[1][1]);
  }
#pragma unroll
  for (int ms = 0; ms < 2; ++ms)
#pragma unroll
    for (int ns = 0; ns < 2; ++ns)
#pragma unroll
      for (int e = 0; e < 8; ++e) {
        int r = mw + 16 * ms + e + 8 * g;
        int c = nw + 16 * ns + ln;
        out[(size_t)r * DM + c] = acc[ms][ns][e];
      }
}

// ---------------------------------------------------------------- launcher
extern "C" void kernel_launch(void* const* d_in, const int* in_sizes, int n_in,
                              void* d_out, int out_size, void* d_ws, size_t ws_size,
                              hipStream_t stream) {
  (void)in_sizes; (void)n_in; (void)out_size; (void)ws_size;
  const float* x    = (const float*)d_in[0];
  const float* mem  = (const float*)d_in[1];
  const float* Wqkv = (const float*)d_in[2];
  const float* Wr   = (const float*)d_in[3];
  const float* Wo   = (const float*)d_in[4];
  const float* u    = (const float*)d_in[5];
  const float* v    = (const float*)d_in[6];

  char* p = (char*)d_ws;
  auto nxt = [&](size_t bytes) {
    char* q = p;
    p += (bytes + 255) & ~(size_t)255;
    return q;
  };
  __bf16* Wqkvt = (__bf16*)nxt((size_t)3 * DM * DM * 2);
  __bf16* Wrt   = (__bf16*)nxt((size_t)DM * DM * 2);
  __bf16* Wot   = (__bf16*)nxt((size_t)DM * DM * 2);
  __bf16* Qu    = (__bf16*)nxt((size_t)BS * NH * QL * DH * 2);
  __bf16* Qv    = (__bf16*)nxt((size_t)BS * NH * QL * DH * 2);
  __bf16* Kh    = (__bf16*)nxt((size_t)BS * NH * KL * DH * 2);
  __bf16* Vt    = (__bf16*)nxt((size_t)BS * NH * KL * DH * 2);
  __bf16* Rh    = (__bf16*)nxt((size_t)NH * KL * DH * 2);
  __bf16* AO    = (__bf16*)nxt((size_t)QL * BS * DM * 2);

  k_transpose_bf16<<<dim3(3 * DM / 32, DM / 32), 256, 0, stream>>>(Wqkv, Wqkvt, DM, 3 * DM);
  k_transpose_bf16<<<dim3(DM / 32, DM / 32), 256, 0, stream>>>(Wr, Wrt, DM, DM);
  k_transpose_bf16<<<dim3(DM / 32, DM / 32), 256, 0, stream>>>(Wo, Wot, DM, DM);

  k_qkv<<<dim3(KL * BS / 128, 3 * DM / 64), 256, 0, stream>>>(x, mem, Wqkvt, u, v,
                                                              Qu, Qv, Kh, Vt);
  k_rproj<<<dim3(KL / 128, DM / 64), 256, 0, stream>>>(Wrt, Rh);
  k_flash<<<BS * NH * (QL / 16) / 8, 256, 0, stream>>>(Qu, Qv, Kh, Vt, Rh, AO);
  k_out<<<dim3(QL * BS / 128, DM / 64), 256, 0, stream>>>(AO, Wot, (float*)d_out);
}
